// WindowAttention3D_24455543783512
// MI455X (gfx1250) — matrix operand order repaired and checked by reading, not compiled, not run on hardware
//
#include <hip/hip_runtime.h>
#include <hip/hip_bf16.h>

typedef __attribute__((ext_vector_type(16))) _Float16 v16h;
typedef __attribute__((ext_vector_type(8)))  _Float16 v8h;
typedef __attribute__((ext_vector_type(4)))  _Float16 v4h;
typedef __attribute__((ext_vector_type(8)))  float    v8f;

#define NTOK 98
#define NPAD 112            // 7*16 token tiles
#define CDIM 96
#define NHEAD 3
#define HDIM 32
#define MT 7
#define QK_SCALE 0.17677669529663688f   // 32^-0.5

// ---- LDS layout (byte offsets) ----
#define SM_QH 0                 // [H][NPAD][HD] f16 = 21504
#define SM_KH 21504             // [H][NPAD][HD] f16 = 21504
#define SM_VT 43008             // [H][HD][128]  f16 = 24576 (cols padded to 128, 112..127 zero)
#define SM_OH 67584             // [NPAD][C]     f16 = 21504
#define SM_XP 89088             // xh [NPAD][C] f16 (21504) ALIASED WITH probs [H][NPAD][128] f16 (86016)
#define SM_TOTAL (89088 + 86016)

// ---- workspace layout (byte offsets) ----
#define WS_QW 0                         // 288*96 f16 = 55296
#define WS_PW 55296                     // 96*96  f16 = 18432
#define WS_BIAS (55296 + 18432)         // 3*112*112 f32 = 150528 (masked pad = -1e30)

__device__ __forceinline__ v8f wmma_f16(v16h a, v16h b, v8f c) {
  return __builtin_amdgcn_wmma_f32_16x16x32_f16(false, a, false, b, (short)0, c, false, false);
}

// A fragment 16x32 f16 from row-major [rows][ld]; rows r0..r0+15, K k0..k0+31.
__device__ __forceinline__ v16h load_a(const _Float16* base, int ld, int r0, int k0, int lane) {
  const _Float16* p = base + (r0 + (lane & 15)) * ld + k0 + ((lane >> 4) << 4);
  return *(const v16h*)p;          // K koff .. koff+15
}

// B fragment 32x16 f16 where memory holds B^T row-major: wt[N][K], stride ld.
__device__ __forceinline__ v16h load_bt(const _Float16* wt, int ld, int n0, int k0, int lane) {
  return *(const v16h*)(wt + (n0 + (lane & 15)) * ld + k0 + ((lane >> 4) << 4));
}

__device__ __forceinline__ float red_max16(float v) {
  v = fmaxf(v, __shfl_xor(v, 1));
  v = fmaxf(v, __shfl_xor(v, 2));
  v = fmaxf(v, __shfl_xor(v, 4));
  v = fmaxf(v, __shfl_xor(v, 8));
  return v;
}
__device__ __forceinline__ float red_sum16(float v) {
  v += __shfl_xor(v, 1);
  v += __shfl_xor(v, 2);
  v += __shfl_xor(v, 4);
  v += __shfl_xor(v, 8);
  return v;
}

// ---------------- pre-pass kernels ----------------
__global__ void prep_weights(const float* __restrict__ qkv_w, const float* __restrict__ proj_w,
                             _Float16* __restrict__ qw, _Float16* __restrict__ pw) {
  int i = blockIdx.x * blockDim.x + threadIdx.x;
  if (i < 288 * CDIM) qw[i] = (_Float16)qkv_w[i];
  if (i < CDIM * CDIM) pw[i] = (_Float16)proj_w[i];
}

__global__ void prep_bias(const float* __restrict__ table, const int* __restrict__ rel,
                          float* __restrict__ bp) {
  int i = blockIdx.x * blockDim.x + threadIdx.x;
  if (i >= NHEAD * NPAD * NPAD) return;
  int h = i / (NPAD * NPAD);
  int r = (i / NPAD) % NPAD;
  int m = i % NPAD;
  float v = -1e30f;  // mask: padded rows/cols vanish through softmax
  if (r < NTOK && m < NTOK) v = table[rel[r * NTOK + m] * NHEAD + h];
  bp[i] = v;
}

// ---------------- fused window-attention kernel ----------------
__global__ void __launch_bounds__(256)
win_attn_kernel(const float* __restrict__ x, const float* __restrict__ qkv_b,
                const float* __restrict__ proj_b, const _Float16* __restrict__ qkv_wh,
                const _Float16* __restrict__ proj_wh, const float* __restrict__ biasPad,
                float* __restrict__ out) {
  extern __shared__ char smem[];
  _Float16* qh     = (_Float16*)(smem + SM_QH);
  _Float16* kh     = (_Float16*)(smem + SM_KH);
  _Float16* vT     = (_Float16*)(smem + SM_VT);
  _Float16* oh     = (_Float16*)(smem + SM_OH);
  _Float16* xh     = (_Float16*)(smem + SM_XP);
  _Float16* probs3 = (_Float16*)(smem + SM_XP);  // [NHEAD][NPAD][128]

  const int b    = blockIdx.x;
  const int tid  = threadIdx.x;
  const int wave = tid >> 5;   // 0..7
  const int lane = tid & 31;
  const int col  = lane & 15;
  const int half = lane >> 4;

  // ---- Phase A: stage x -> LDS f16 (rows >=98 zero); zero vT K-pad cols ----
  const float4* xs = (const float4*)(x + (size_t)b * NTOK * CDIM);
  for (int g = tid; g < NPAD * (CDIM / 4); g += 256) {
    int row = g / (CDIM / 4), c4 = g % (CDIM / 4);
    float4 f = (row < NTOK) ? xs[row * (CDIM / 4) + c4] : make_float4(0.f, 0.f, 0.f, 0.f);
    v4h h4 = { (_Float16)f.x, (_Float16)f.y, (_Float16)f.z, (_Float16)f.w };
    *(v4h*)(xh + row * CDIM + c4 * 4) = h4;
  }
  for (int g = tid; g < NHEAD * HDIM * 16; g += 256)
    vT[(g >> 4) * 128 + 112 + (g & 15)] = (_Float16)0.f;
  __syncthreads();

  // ---- Phase B: qkv = x @ qkv_w^T + b -> q(scaled)/k/vT. Three uniform loops so the
  //      q/k/v routing is a compile-time constant (no lane-predicated stores). ----
  // B.q
  for (int t = wave; t < MT * 6; t += 8) {
    int mt = t / 6, ot = t % 6;
    int o0 = ot * 16;                 // 0..95
    int head = o0 >> 5, dbase = o0 & 31;
    v8f acc = {};
#pragma unroll
    for (int kk = 0; kk < 3; ++kk) {
      v16h a   = load_a(xh, CDIM, mt * 16, kk * 32, lane);
      v16h bfr = load_bt(qkv_wh, CDIM, o0, kk * 32, lane);
      acc = wmma_f16(a, bfr, acc);
    }
    float bc = qkv_b[o0 + col];
#pragma unroll
    for (int vi = 0; vi < 8; ++vi) {
      int M = mt * 16 + vi + half * 8;
      qh[(head * NPAD + M) * HDIM + dbase + col] = (_Float16)((acc[vi] + bc) * QK_SCALE);
    }
  }
  // B.k
  for (int t = wave; t < MT * 6; t += 8) {
    int mt = t / 6, ot = t % 6;
    int o0 = 96 + ot * 16;
    int rem = o0 - 96;
    int head = rem >> 5, dbase = rem & 31;
    v8f acc = {};
#pragma unroll
    for (int kk = 0; kk < 3; ++kk) {
      v16h a   = load_a(xh, CDIM, mt * 16, kk * 32, lane);
      v16h bfr = load_bt(qkv_wh, CDIM, o0, kk * 32, lane);
      acc = wmma_f16(a, bfr, acc);
    }
    float bc = qkv_b[o0 + col];
#pragma unroll
    for (int vi = 0; vi < 8; ++vi) {
      int M = mt * 16 + vi + half * 8;
      kh[(head * NPAD + M) * HDIM + dbase + col] = (_Float16)(acc[vi] + bc);
    }
  }
  // B.v (stored transposed: vT[h][d][token], token dim padded to 128)
  for (int t = wave; t < MT * 6; t += 8) {
    int mt = t / 6, ot = t % 6;
    int o0 = 192 + ot * 16;
    int rem = o0 - 192;
    int head = rem >> 5, dbase = rem & 31;
    v8f acc = {};
#pragma unroll
    for (int kk = 0; kk < 3; ++kk) {
      v16h a   = load_a(xh, CDIM, mt * 16, kk * 32, lane);
      v16h bfr = load_bt(qkv_wh, CDIM, o0, kk * 32, lane);
      acc = wmma_f16(a, bfr, acc);
    }
    float bc = qkv_b[o0 + col];
#pragma unroll
    for (int vi = 0; vi < 8; ++vi) {
      int M = mt * 16 + vi + half * 8;
      vT[(head * HDIM + dbase + col) * 128 + M] = (_Float16)(acc[vi] + bc);
    }
  }
  __syncthreads();

  // ---- Phase C1: all heads' scores + softmax -> probs3 (21 jobs over 8 waves) ----
  // zero the K-pad columns (112..127) of every probs row, cooperatively
  for (int g = tid; g < NHEAD * NPAD * 16; g += 256)
    probs3[(g >> 4) * 128 + 112 + (g & 15)] = (_Float16)0.f;

  for (int t = wave; t < NHEAD * MT; t += 8) {
    int h = t / MT, nt = t % MT;
    const _Float16* qhh = qh + h * NPAD * HDIM;
    const _Float16* khh = kh + h * NPAD * HDIM;
    _Float16* probs = probs3 + h * NPAD * 128;

    v16h a = load_a(qhh, HDIM, nt * 16, 0, lane);
    v8f s[MT];
#pragma unroll
    for (int mt2 = 0; mt2 < MT; ++mt2) {
      v16h bfr = load_bt(khh, HDIM, mt2 * 16, 0, lane);
      v8f z = {};
      s[mt2] = wmma_f16(a, bfr, z);
    }
#pragma unroll
    for (int mt2 = 0; mt2 < MT; ++mt2)
#pragma unroll
      for (int vi = 0; vi < 8; ++vi) {
        int n = nt * 16 + vi + half * 8;
        s[mt2][vi] += biasPad[(h * NPAD + n) * NPAD + mt2 * 16 + col];
      }
#pragma unroll
    for (int vi = 0; vi < 8; ++vi) {
      float mx = s[0][vi];
#pragma unroll
      for (int mt2 = 1; mt2 < MT; ++mt2) mx = fmaxf(mx, s[mt2][vi]);
      mx = red_max16(mx);
      float sum = 0.f;
#pragma unroll
      for (int mt2 = 0; mt2 < MT; ++mt2) {
        float e = __expf(s[mt2][vi] - mx);
        s[mt2][vi] = e;
        sum += e;
      }
      sum = red_sum16(sum);
      float inv = 1.0f / sum;
      int row = nt * 16 + vi + half * 8;
#pragma unroll
      for (int mt2 = 0; mt2 < MT; ++mt2)
        probs[row * 128 + mt2 * 16 + col] = (_Float16)(s[mt2][vi] * inv);
    }
  }
  __syncthreads();

  // ---- Phase C2: out_h = probs @ v for all heads (42 jobs over 8 waves) ----
  for (int t = wave; t < NHEAD * MT * 2; t += 8) {
    int h = t / (MT * 2);
    int r = t % (MT * 2);
    int nt = r >> 1, dt = r & 1;
    const _Float16* probs = probs3 + h * NPAD * 128;
    const _Float16* vth   = vT + h * HDIM * 128;
    v8f acc = {};
#pragma unroll
    for (int km = 0; km < 4; ++km) {
      v16h a   = load_a(probs, 128, nt * 16, km * 32, lane);
      v16h bfr = load_bt(vth, 128, dt * 16, km * 32, lane);
      acc = wmma_f16(a, bfr, acc);
    }
#pragma unroll
    for (int vi = 0; vi < 8; ++vi) {
      int M = nt * 16 + vi + half * 8;
      oh[M * CDIM + h * HDIM + dt * 16 + col] = (_Float16)acc[vi];
    }
  }
  __syncthreads();

  // ---- Phase D: out = oh @ proj_w^T + proj_b (f32 store) ----
  for (int t = wave; t < MT * 6; t += 8) {
    int nt = t / 6, ot = t % 6;
    int o0 = ot * 16;
    v8f acc = {};
#pragma unroll
    for (int kk = 0; kk < 3; ++kk) {
      v16h a   = load_a(oh, CDIM, nt * 16, kk * 32, lane);
      v16h bfr = load_bt(proj_wh, CDIM, o0, kk * 32, lane);
      acc = wmma_f16(a, bfr, acc);
    }
    float bc = proj_b[o0 + col];
#pragma unroll
    for (int vi = 0; vi < 8; ++vi) {
      int n = nt * 16 + vi + half * 8;
      if (n < NTOK)
        out[((size_t)b * NTOK + n) * CDIM + o0 + col] = acc[vi] + bc;
    }
  }
}

extern "C" void kernel_launch(void* const* d_in, const int* in_sizes, int n_in,
                              void* d_out, int out_size, void* d_ws, size_t ws_size,
                              hipStream_t stream) {
  const float* x          = (const float*)d_in[0];
  const float* qkv_w      = (const float*)d_in[1];
  const float* qkv_b      = (const float*)d_in[2];
  const float* proj_w     = (const float*)d_in[3];
  const float* proj_b     = (const float*)d_in[4];
  const float* bias_table = (const float*)d_in[5];
  const int*   rel_index  = (const int*)d_in[6];
  float* out = (float*)d_out;

  _Float16* qkv_wh  = (_Float16*)((char*)d_ws + WS_QW);
  _Float16* proj_wh = (_Float16*)((char*)d_ws + WS_PW);
  float*    biasPad = (float*)((char*)d_ws + WS_BIAS);

  prep_weights<<<(288 * CDIM + 255) / 256, 256, 0, stream>>>(qkv_w, proj_w, qkv_wh, proj_wh);
  prep_bias<<<(NHEAD * NPAD * NPAD + 255) / 256, 256, 0, stream>>>(bias_table, rel_index, biasPad);

  int nwin = in_sizes[0] / (NTOK * CDIM);  // 4096
  win_attn_kernel<<<nwin, 256, SM_TOTAL, stream>>>(x, qkv_b, proj_b, qkv_wh, proj_wh, biasPad, out);
}
